// SimpleTransformerLM_36232344109736
// MI455X (gfx1250) — compile-verified, matrix operand order checked by reading, not run on hardware
//
#include <hip/hip_runtime.h>

// ---------------------------------------------------------------------------
// SimpleTransformerLM forward for gfx1250 (MI455X)
//  - all GEMM operands pre-converted to bf16, f32 WMMA accumulate
//  - GEMM stages tiles with global_load_async_to_lds_b128 + double buffering
//  - flash attention on bf16 Q/K/V with WMMA for QK^T and P*V
// ---------------------------------------------------------------------------

#define BB 2
#define SS 1024
#define DD 1024
#define HH 16
#define LL 4
#define FF_ 4096
#define VV 32000
#define DKK 64

typedef __attribute__((ext_vector_type(16))) __bf16 v16bf;
typedef __attribute__((ext_vector_type(8)))  float  v8f;

union Frag  { uint4 u[2]; v16bf v; };
union U4S8  { uint4 u; unsigned short s[8]; };

__device__ __forceinline__ unsigned short f2bfu(float f) {
  union { float f; unsigned u; } x; x.f = f;
  unsigned r = x.u + 0x7FFFu + ((x.u >> 16) & 1u);   // round-to-nearest-even
  return (unsigned short)(r >> 16);
}
__device__ __forceinline__ unsigned pk2(float a, float b) {
  return (unsigned)f2bfu(a) | ((unsigned)f2bfu(b) << 16);
}
__device__ __forceinline__ v8f wmma_bf16(const Frag& a, const Frag& b, v8f c) {
  return __builtin_amdgcn_wmma_f32_16x16x32_bf16(
      false, a.v, false, b.v, (short)0, c, false, false);
}
__device__ __forceinline__ float redmax16(float v) {
  #pragma unroll
  for (int m = 8; m >= 1; m >>= 1) v = fmaxf(v, __shfl_xor(v, m, 32));
  return v;
}
__device__ __forceinline__ float redsum16(float v) {
  #pragma unroll
  for (int m = 8; m >= 1; m >>= 1) v += __shfl_xor(v, m, 32);
  return v;
}

// ---- CDNA5 async global->LDS copy (ASYNCcnt path) --------------------------
__device__ __forceinline__ unsigned lds_off32(const void* p) {
  return (unsigned)(unsigned long long)p;   // flat LDS addr: low 32 bits = LDS offset
}
__device__ __forceinline__ void async_b128(unsigned ldsoff, const void* gaddr) {
  asm volatile("global_load_async_to_lds_b128 %0, %1, off"
               :: "v"(ldsoff), "v"(gaddr) : "memory");
}
__device__ __forceinline__ void wait_async0() {
  asm volatile("s_wait_asynccnt 0x0" ::: "memory");
}

// ---------------------------------------------------------------------------
// fp32 -> bf16 conversion (weights / staging), 8 elems per thread
// ---------------------------------------------------------------------------
__global__ __launch_bounds__(256)
void stlm_f32_to_bf16(const float* __restrict__ src,
                      unsigned short* __restrict__ dst) {
  const long long i = ((long long)blockIdx.x * 256 + threadIdx.x) * 8;
  float4 f0 = *reinterpret_cast<const float4*>(src + i);
  float4 f1 = *reinterpret_cast<const float4*>(src + i + 4);
  uint4 o;
  o.x = pk2(f0.x, f0.y); o.y = pk2(f0.z, f0.w);
  o.z = pk2(f1.x, f1.y); o.w = pk2(f1.z, f1.w);
  *reinterpret_cast<uint4*>(dst + i) = o;
}

// ---------------------------------------------------------------------------
// Embedding + sinusoidal positional encoding -> fp32 X and bf16 Xb
// ---------------------------------------------------------------------------
__global__ __launch_bounds__(256)
void stlm_embed_pe(const int* __restrict__ ids, const float* __restrict__ emb,
                   float* __restrict__ x, unsigned short* __restrict__ xb) {
  const int bs = blockIdx.x;             // b*S + s
  const int s  = bs & (SS - 1);
  const int id = ids[bs];
  const float sqrtD = 32.0f;             // sqrt(1024)
  const float c = -logf(10000.0f) / (float)DD;
  #pragma unroll
  for (int j = 0; j < 4; ++j) {
    const int d  = threadIdx.x * 4 + j;
    const int jj = d >> 1;
    const float ang = (float)s * __expf(c * (float)(2 * jj));
    const float pe  = (d & 1) ? __cosf(ang) : __sinf(ang);
    const float v   = emb[(size_t)id * DD + d] * sqrtD + pe;
    x[(size_t)bs * DD + d]  = v;
    xb[(size_t)bs * DD + d] = f2bfu(v);
  }
}

// ---------------------------------------------------------------------------
// GEMM: C[M,N] = A[M,K] @ W[N,K]^T + bias  (optional ReLU)
// A, W bf16; outputs optional fp32 C and/or bf16 Cb.
// 256 thr = 8 waves; block tile 128x128; wave tile 32x64; K-step 32.
// Tiles staged with async global->LDS copies, double-buffered.
// ---------------------------------------------------------------------------
#define GSTR 40   // LDS row stride (bf16 elems) for 32-wide K tiles

__global__ __launch_bounds__(256)
void stlm_gemm_bf(const unsigned short* __restrict__ A,
                  const unsigned short* __restrict__ W,
                  const float* __restrict__ bias,
                  float* __restrict__ C, unsigned short* __restrict__ Cb,
                  int M, int N, int K, int relu) {
  __shared__ unsigned short As[2][128 * GSTR];
  __shared__ unsigned short Bs[2][128 * GSTR];

  const int tid    = threadIdx.x;
  const int lane   = tid & 31;
  const int wave   = tid >> 5;
  const int waveM  = wave >> 1;          // 0..3
  const int waveN  = wave & 1;           // 0..1
  const int lanelo = lane & 15;
  const int lanehi = lane >> 4;
  const int blockM = blockIdx.y * 128;
  const int blockN = blockIdx.x * 128;
  const int kb0    = lanehi * 8;         // per-lane K base inside 32-chunk

  v8f acc[2][4];
  #pragma unroll
  for (int i = 0; i < 2; ++i)
    #pragma unroll
    for (int j = 0; j < 4; ++j) acc[i][j] = (v8f)0.0f;

  const int ldrow = tid >> 1;            // 0..127
  const int ldcol = (tid & 1) * 16;      // 0 or 16

  // issue one 128x32 bf16 tile per operand into LDS buffer `buf`
  auto issue = [&](int buf, int k0) {
    const unsigned short* ag = A + (size_t)(blockM + ldrow) * K + k0 + ldcol;
    const unsigned short* bg = W + (size_t)(blockN + ldrow) * K + k0 + ldcol;
    const unsigned aoff = lds_off32(&As[buf][ldrow * GSTR + ldcol]);
    const unsigned boff = lds_off32(&Bs[buf][ldrow * GSTR + ldcol]);
    async_b128(aoff,      ag);
    async_b128(aoff + 16, ag + 8);
    async_b128(boff,      bg);
    async_b128(boff + 16, bg + 8);
  };

  const int nk = K / 32;
  issue(0, 0);

  for (int i = 0; i < nk; ++i) {
    const int cur = i & 1;
    wait_async0();                       // this iteration's tiles landed
    __syncthreads();                     // visible to all waves; prev reads done
    if (i + 1 < nk) issue(cur ^ 1, (i + 1) * 32);

    Frag af[2], bf[4];
    #pragma unroll
    for (int wm = 0; wm < 2; ++wm) {
      const int r = waveM * 32 + wm * 16 + lanelo;
      af[wm].u[0] = *reinterpret_cast<const uint4*>(&As[cur][r * GSTR + kb0]);
      af[wm].u[1] = *reinterpret_cast<const uint4*>(&As[cur][r * GSTR + kb0 + 16]);
    }
    #pragma unroll
    for (int wn = 0; wn < 4; ++wn) {
      const int r = waveN * 64 + wn * 16 + lanelo;
      bf[wn].u[0] = *reinterpret_cast<const uint4*>(&Bs[cur][r * GSTR + kb0]);
      bf[wn].u[1] = *reinterpret_cast<const uint4*>(&Bs[cur][r * GSTR + kb0 + 16]);
    }
    #pragma unroll
    for (int wm = 0; wm < 2; ++wm)
      #pragma unroll
      for (int wn = 0; wn < 4; ++wn)
        acc[wm][wn] = wmma_bf16(af[wm], bf[wn], acc[wm][wn]);
    __syncthreads();                     // done reading `cur` before refill
  }

  // ---- epilogue ----
  #pragma unroll
  for (int wn = 0; wn < 4; ++wn) {
    const int col = blockN + waveN * 64 + wn * 16 + lanelo;
    const float bv = bias ? bias[col] : 0.0f;
    #pragma unroll
    for (int wm = 0; wm < 2; ++wm) {
      #pragma unroll
      for (int r = 0; r < 8; ++r) {
        const int row = blockM + waveM * 32 + wm * 16 + r + lanehi * 8;
        float v = acc[wm][wn][r] + bv;
        if (relu) v = fmaxf(v, 0.0f);
        if (C)  C[(size_t)row * N + col]  = v;
        if (Cb) Cb[(size_t)row * N + col] = f2bfu(v);
      }
    }
  }
}

// ---------------------------------------------------------------------------
// Flash-style causal attention on bf16 Q/K/V -> bf16 O
// one workgroup per (b, h, 64 q-rows); 4 waves x 16 q-rows
// K tiles staged via async global->LDS; V transposed through VGPRs
// ---------------------------------------------------------------------------
#define KSTR 72   // Ktile LDS row stride (bf16), row = key, 64 dk
#define VSTR 40   // VT   LDS row stride (bf16), row = dk,  32 keys
#define PSTR 40   // P    LDS row stride (bf16), row = q,   32 keys

__global__ __launch_bounds__(128)
void stlm_attention(const unsigned short* __restrict__ Q,
                    const unsigned short* __restrict__ Km,
                    const unsigned short* __restrict__ Vm,
                    unsigned short* __restrict__ O, float scale) {
  __shared__ unsigned short Ks[32 * KSTR];
  __shared__ unsigned short Vs[64 * VSTR];        // transposed: [dk][key]
  __shared__ unsigned short Ps[4][16 * PSTR];

  const int qt   = blockIdx.x;                    // 0..S/64-1
  const int h    = blockIdx.y;
  const int b    = blockIdx.z;
  const int tid  = threadIdx.x;
  const int lane = tid & 31;
  const int wave = tid >> 5;
  const int lanelo = lane & 15;
  const int lanehi = lane >> 4;
  const int qbase  = qt * 64 + wave * 16;
  const int kb0    = lanehi * 8;

  // ---- Q fragments (rows qbase..+15, dk 0..63), direct bf16 loads ----
  const unsigned short* qp =
      Q + ((size_t)(b * SS + qbase + lanelo)) * DD + h * DKK;
  Frag a0, a1;
  a0.u[0] = *reinterpret_cast<const uint4*>(qp + kb0);
  a0.u[1] = *reinterpret_cast<const uint4*>(qp + kb0 + 16);
  a1.u[0] = *reinterpret_cast<const uint4*>(qp + kb0 + 32);
  a1.u[1] = *reinterpret_cast<const uint4*>(qp + kb0 + 48);

  float mold[8], lsum[8];
  v8f oacc[4];
  #pragma unroll
  for (int r = 0; r < 8; ++r) { mold[r] = -1e30f; lsum[r] = 0.0f; }
  #pragma unroll
  for (int g = 0; g < 4; ++g) oacc[g] = (v8f)0.0f;

  const int ntiles = qt * 2 + 2;                   // keys [0, qt*64+64) in 32s
  const int srow = tid >> 2;                       // staging: key row 0..31
  const int scol = (tid & 3) * 16;                 // staging: dk col base

  for (int kt = 0; kt < ntiles; ++kt) {
    const int kb = kt * 32;

    // ---- K tile: async global -> LDS (row-major [key][dk]) ----
    const unsigned short* kg =
        Km + ((size_t)(b * SS + kb + srow)) * DD + h * DKK + scol;
    const unsigned koff = lds_off32(&Ks[srow * KSTR + scol]);
    async_b128(koff,      kg);
    async_b128(koff + 16, kg + 8);

    // ---- V tile: load through VGPRs, store transposed [dk][key] ----
    const unsigned short* vg =
        Vm + ((size_t)(b * SS + kb + srow)) * DD + h * DKK + scol;
    U4S8 v0, v1;
    v0.u = *reinterpret_cast<const uint4*>(vg);
    v1.u = *reinterpret_cast<const uint4*>(vg + 8);
    #pragma unroll
    for (int j = 0; j < 8; ++j) {
      Vs[(scol + j) * VSTR + srow]     = v0.s[j];
      Vs[(scol + 8 + j) * VSTR + srow] = v1.s[j];
    }
    wait_async0();
    __syncthreads();

    // ---- scores: two 16-key halves, each = Q[16x64] x K^T chunk ----
    v8f sc[2];
    #pragma unroll
    for (int half = 0; half < 2; ++half) {
      const int krow = half * 16 + lanelo;
      Frag bk0, bk1;
      bk0.u[0] = *reinterpret_cast<const uint4*>(&Ks[krow * KSTR + kb0]);
      bk0.u[1] = *reinterpret_cast<const uint4*>(&Ks[krow * KSTR + kb0 + 16]);
      bk1.u[0] = *reinterpret_cast<const uint4*>(&Ks[krow * KSTR + kb0 + 32]);
      bk1.u[1] = *reinterpret_cast<const uint4*>(&Ks[krow * KSTR + kb0 + 48]);
      v8f s = (v8f)0.0f;
      s = wmma_bf16(a0, bk0, s);
      s = wmma_bf16(a1, bk1, s);
      sc[half] = s;
    }

    // ---- online softmax update ----
    unsigned short* Pl = &Ps[wave][0];
    #pragma unroll
    for (int r = 0; r < 8; ++r) {
      const int m    = r + lanehi * 8;
      const int qrow = qbase + m;
      const int key0 = kb + lanelo;
      const int key1 = kb + 16 + lanelo;
      float s0 = (key0 <= qrow) ? sc[0][r] * scale : -1e30f;
      float s1 = (key1 <= qrow) ? sc[1][r] * scale : -1e30f;
      const float rm   = redmax16(fmaxf(s0, s1));
      const float mnew = fmaxf(mold[r], rm);
      const float alph = __expf(mold[r] - mnew);
      const float p0   = __expf(s0 - mnew);
      const float p1   = __expf(s1 - mnew);
      const float rs   = redsum16(p0 + p1);
      lsum[r] = lsum[r] * alph + rs;
      mold[r] = mnew;
      #pragma unroll
      for (int g = 0; g < 4; ++g) oacc[g][r] *= alph;
      Pl[m * PSTR + lanelo]      = f2bfu(p0);
      Pl[m * PSTR + 16 + lanelo] = f2bfu(p1);
    }

    // ---- O += P[16x32] x V[32x64] ----
    Frag aP;
    aP.u[0] = *reinterpret_cast<const uint4*>(&Pl[lanelo * PSTR + kb0]);
    aP.u[1] = *reinterpret_cast<const uint4*>(&Pl[lanelo * PSTR + kb0 + 16]);
    #pragma unroll
    for (int g = 0; g < 4; ++g) {
      const int vrow = g * 16 + lanelo;               // dk column
      Frag bv;
      bv.u[0] = *reinterpret_cast<const uint4*>(&Vs[vrow * VSTR + kb0]);
      bv.u[1] = *reinterpret_cast<const uint4*>(&Vs[vrow * VSTR + kb0 + 16]);
      oacc[g] = wmma_bf16(aP, bv, oacc[g]);
    }
    __syncthreads();
  }

  // ---- normalize and store bf16 ----
  #pragma unroll
  for (int g = 0; g < 4; ++g) {
    #pragma unroll
    for (int r = 0; r < 8; ++r) {
      const int m = r + lanehi * 8;
      const int s = qbase + m;
      O[((size_t)(b * SS + s)) * DD + h * DKK + g * 16 + lanelo] =
          f2bfu(oacc[g][r] / lsum[r]);
    }
  }
}

// ---------------------------------------------------------------------------
// out = LayerNorm(x (+ y)) * g + b ; fp32 out + optional bf16 copy
// ---------------------------------------------------------------------------
__global__ __launch_bounds__(256)
void stlm_add_ln(const float* __restrict__ x, const float* __restrict__ y,
                 const float* __restrict__ g, const float* __restrict__ b,
                 float* __restrict__ out, unsigned short* __restrict__ outb) {
  __shared__ float r1[8], r2[8];
  const int row = blockIdx.x;
  const int t   = threadIdx.x;
  const int d0  = t * 4;
  const size_t base = (size_t)row * DD;

  float4 xv = *reinterpret_cast<const float4*>(x + base + d0);
  if (y) {
    float4 yv = *reinterpret_cast<const float4*>(y + base + d0);
    xv.x += yv.x; xv.y += yv.y; xv.z += yv.z; xv.w += yv.w;
  }
  float s = xv.x + xv.y + xv.z + xv.w;
  #pragma unroll
  for (int m = 16; m >= 1; m >>= 1) s += __shfl_xor(s, m, 32);
  if ((t & 31) == 0) r1[t >> 5] = s;
  __syncthreads();
  float tot = 0.0f;
  #pragma unroll
  for (int i = 0; i < 8; ++i) tot += r1[i];
  const float mean = tot * (1.0f / (float)DD);

  const float dx0 = xv.x - mean, dx1 = xv.y - mean;
  const float dx2 = xv.z - mean, dx3 = xv.w - mean;
  float ss = dx0 * dx0 + dx1 * dx1 + dx2 * dx2 + dx3 * dx3;
  #pragma unroll
  for (int m = 16; m >= 1; m >>= 1) ss += __shfl_xor(ss, m, 32);
  if ((t & 31) == 0) r2[t >> 5] = ss;
  __syncthreads();
  float vtot = 0.0f;
  #pragma unroll
  for (int i = 0; i < 8; ++i) vtot += r2[i];
  const float inv = rsqrtf(vtot * (1.0f / (float)DD) + 1e-5f);

  #pragma unroll
  for (int j = 0; j < 4; ++j) {
    const float dx = (&dx0)[0] * 0.0f +   // keep simple: recompute below
                     0.0f;
    (void)dx;
  }
  const float o0 = dx0 * inv * g[d0 + 0] + b[d0 + 0];
  const float o1 = dx1 * inv * g[d0 + 1] + b[d0 + 1];
  const float o2 = dx2 * inv * g[d0 + 2] + b[d0 + 2];
  const float o3 = dx3 * inv * g[d0 + 3] + b[d0 + 3];
  out[base + d0 + 0] = o0;
  out[base + d0 + 1] = o1;
  out[base + d0 + 2] = o2;
  out[base + d0 + 3] = o3;
  if (outb) {
    outb[base + d0 + 0] = f2bfu(o0);
    outb[base + d0 + 1] = f2bfu(o1);
    outb[base + d0 + 2] = f2bfu(o2);
    outb[base + d0 + 3] = f2bfu(o3);
  }
}

// ---------------------------------------------------------------------------
// host launcher
// ---------------------------------------------------------------------------
extern "C" void kernel_launch(void* const* d_in, const int* in_sizes, int n_in,
                              void* d_out, int out_size, void* d_ws, size_t ws_size,
                              hipStream_t stream) {
  (void)in_sizes; (void)n_in; (void)out_size; (void)ws_size;

  const int*   ids   = (const int*)  d_in[0];
  const float* emb   = (const float*)d_in[1];
  const float* Wq    = (const float*)d_in[2];
  const float* bq    = (const float*)d_in[3];
  const float* Wk    = (const float*)d_in[4];
  const float* bk    = (const float*)d_in[5];
  const float* Wv    = (const float*)d_in[6];
  const float* bv    = (const float*)d_in[7];
  const float* Wo    = (const float*)d_in[8];
  const float* bo    = (const float*)d_in[9];
  const float* ln1g  = (const float*)d_in[10];
  const float* ln1b  = (const float*)d_in[11];
  const float* W1    = (const float*)d_in[12];
  const float* b1    = (const float*)d_in[13];
  const float* W2    = (const float*)d_in[14];
  const float* b2    = (const float*)d_in[15];
  const float* ln2g  = (const float*)d_in[16];
  const float* ln2b  = (const float*)d_in[17];
  const float* lnfg  = (const float*)d_in[18];
  const float* lnfb  = (const float*)d_in[19];
  const float* headw = (const float*)d_in[20];

  const int M = BB * SS;                          // 2048 rows
  const size_t nBSD = (size_t)M * DD;             // 2,097,152 elems
  const size_t nBSF = (size_t)M * FF_;            // 8,388,608 elems

  // workspace layout (fp32 first, then bf16 buffers; all 16B-aligned)
  float* X  = (float*)d_ws;                       // [M,D] fp32
  float* Pb = X + nBSD;                           // [M,D] fp32
  unsigned short* Xb  = (unsigned short*)(Pb + nBSD);   // [M,D] bf16
  unsigned short* Qb  = Xb  + nBSD;
  unsigned short* Kb  = Qb  + nBSD;
  unsigned short* Vb  = Kb  + nBSD;
  unsigned short* Ob  = Vb  + nBSD;
  unsigned short* FFb = Ob  + nBSD;               // [M,F] bf16
  unsigned short* Wst = FFb + nBSF;               // weight staging, up to V*D bf16

  auto conv = [&](const float* s, unsigned short* d, size_t n) {
    stlm_f32_to_bf16<<<(unsigned)(n / 2048), 256, 0, stream>>>(s, d);
  };
  auto gemm = [&](const unsigned short* A, const float* Wf, size_t wn,
                  const float* bias, float* C, unsigned short* Cb,
                  int Mm, int Nn, int Kk, int relu) {
    conv(Wf, Wst, wn);
    dim3 grid(Nn / 128, Mm / 128);
    stlm_gemm_bf<<<grid, 256, 0, stream>>>(A, Wst, bias, C, Cb, Mm, Nn, Kk, relu);
  };

  stlm_embed_pe<<<M, 256, 0, stream>>>(ids, emb, X, Xb);

  const float scale = 0.125f;                     // 1/sqrt(64)
  for (int l = 0; l < LL; ++l) {
    const size_t wDD = (size_t)l * DD * DD;
    const size_t wFD = (size_t)l * FF_ * DD;
    gemm(Xb, Wq + wDD, (size_t)DD * DD, bq + l * DD, nullptr, Qb, M, DD, DD, 0);
    gemm(Xb, Wk + wDD, (size_t)DD * DD, bk + l * DD, nullptr, Kb, M, DD, DD, 0);
    gemm(Xb, Wv + wDD, (size_t)DD * DD, bv + l * DD, nullptr, Vb, M, DD, DD, 0);
    stlm_attention<<<dim3(SS / 64, HH, BB), 128, 0, stream>>>(Qb, Kb, Vb, Ob, scale);
    gemm(Ob, Wo + wDD, (size_t)DD * DD, bo + l * DD, Pb, nullptr, M, DD, DD, 0);
    stlm_add_ln<<<M, 256, 0, stream>>>(X, Pb, ln1g + l * DD, ln1b + l * DD, X, Xb);
    gemm(Xb, W1 + wFD, (size_t)FF_ * DD, b1 + l * FF_, nullptr, FFb, M, FF_, DD, 1);
    gemm(FFb, W2 + wFD, (size_t)FF_ * DD, b2 + l * DD, Pb, nullptr, M, DD, FF_, 0);
    stlm_add_ln<<<M, 256, 0, stream>>>(X, Pb, ln2g + l * DD, ln2b + l * DD, X, Xb);
  }

  // final LN: bf16 result into Xb (X fp32 no longer needed afterwards)
  stlm_add_ln<<<M, 256, 0, stream>>>(X, nullptr, lnfg, lnfb, Pb, Xb);
  gemm(Xb, headw, (size_t)VV * DD, nullptr, (float*)d_out, nullptr, M, VV, DD, 0);
}
